// DynamicMoERoutingLayer_40544491274390
// MI455X (gfx1250) — matrix operand order repaired and checked by reading, compiled, not run on hardware
//
#include <hip/hip_runtime.h>
#include <hip/hip_bf16.h>
#include <stdint.h>

// -------- types for CDNA5 WMMA --------
typedef __attribute__((ext_vector_type(16))) __bf16 v16bf;
typedef __attribute__((ext_vector_type(8)))  float  v8f;

#define NB    10
#define BATCH 32
#define CIN   64
#define COUT  64
#define H     64
#define W     64
#define HO    62
#define WO    62
#define RDIM  512
#define DDIM  128

#define FRAGS_PER_B 36864          // 9 pos * 2 kstep * 4 mgroup * 32 lanes * 16 bf16
#define LCOLS  66                  // 64 valid cols + 2 halo for kw
#define LPITCH 72                  // cin pitch (bf16) in LDS: 144B = 9*16B -> conflict-friendly
#define LDSX_BYTES (6 * LCOLS * LPITCH * 2)

__device__ __forceinline__ unsigned short f2bf(float f) {
  unsigned int u = __float_as_uint(f);
  unsigned int r = u + 0x7FFFu + ((u >> 16) & 1u);   // round-to-nearest-even
  return (unsigned short)(r >> 16);
}

// ============================================================================
// Kernel 1: routing projection + cosine sim + softmax -> wroute[b][10],
//           plus effective bias bias_eff[b][64] = sum_n w_n * conv_b[n]
// ============================================================================
__global__ void routing_kernel(const float* __restrict__ rv,
                               const float* __restrict__ rp_w,
                               const float* __restrict__ rp_b,
                               const float* __restrict__ emb,
                               const float* __restrict__ conv_b,
                               float* __restrict__ wroute,
                               float* __restrict__ bias_eff) {
  const int b   = blockIdx.x;
  const int tid = threadIdx.x;                 // 128 threads
  __shared__ float sr[DDIM];
  __shared__ float snorm;
  __shared__ float sw[NB];

  // r[b, tid] = rv[b,:] . rp_w[tid,:] + rp_b[tid]
  float acc = rp_b[tid];
  const float* rvb  = rv + (size_t)b * RDIM;
  const float* wrow = rp_w + (size_t)tid * RDIM;
  for (int k = 0; k < RDIM; ++k) acc = fmaf(rvb[k], wrow[k], acc);
  sr[tid] = acc;
  __syncthreads();

  if (tid == 0) {
    float s = 0.f;
    for (int i = 0; i < DDIM; ++i) s += sr[i] * sr[i];
    snorm = fmaxf(sqrtf(s), 1e-8f);
  }
  __syncthreads();

  if (tid < NB) {
    float dot = 0.f, en = 0.f;
    const float* e = emb + (size_t)tid * DDIM;
    for (int d = 0; d < DDIM; ++d) { dot = fmaf(sr[d], e[d], dot); en = fmaf(e[d], e[d], en); }
    sw[tid] = dot / (snorm * fmaxf(sqrtf(en), 1e-8f));
  }
  __syncthreads();

  if (tid == 0) {
    float m = sw[0];
    for (int n = 1; n < NB; ++n) m = fmaxf(m, sw[n]);
    float s = 0.f;
    for (int n = 0; n < NB; ++n) { sw[n] = __expf(sw[n] - m); s += sw[n]; }
    float d = 0.f;
    for (int n = 0; n < NB; ++n) { sw[n] /= s; d += sw[n]; }     // softmax
    for (int n = 0; n < NB; ++n) { sw[n] /= d; wroute[b * NB + n] = sw[n]; } // fold /sum(w)
  }
  __syncthreads();

  if (tid < COUT) {
    float be = 0.f;
    for (int n = 0; n < NB; ++n) be = fmaf(sw[n], conv_b[n * COUT + tid], be);
    bias_eff[b * COUT + tid] = be;
  }
}

// ============================================================================
// Kernel 2: per-sample effective weights, emitted directly in the CDNA5
// 16-bit A-matrix (16x32) WMMA layout but with contiguous k (not the ISA table order; B matches), as bf16:
//   wmix[b][pos(9)][ks(2)][mg(4)][lane(32)][e(16)]
//   lane = half*16 + m (cout = mg*16+m); k_local = e + 16*half
// ============================================================================
__global__ void mixw_kernel(const float* __restrict__ conv_w,
                            const float* __restrict__ wroute,
                            unsigned short* __restrict__ wmix) {
  const int gid = blockIdx.x * blockDim.x + threadIdx.x;   // < 32*36864
  const int b = gid / FRAGS_PER_B;
  const int t = gid % FRAGS_PER_B;
  const int e    = t & 15;
  const int lane = (t >> 4) & 31;
  const int mg   = (t >> 9) & 3;
  const int ks   = (t >> 11) & 1;
  const int pos  = t >> 12;                                // 0..8
  const int kh = pos / 3, kw = pos % 3;
  const int m = lane & 15, half = lane >> 4;
  const int cout = mg * 16 + m;
  const int kl   = e + 16 * half;
  const int cin  = ks * 32 + kl;

  const float* wr = wroute + b * NB;
  const int widx = (cout * CIN + cin) * 9 + kh * 3 + kw;   // offset within one expert
  float v = 0.f;
  #pragma unroll
  for (int n = 0; n < NB; ++n)
    v = fmaf(wr[n], conv_w[n * (COUT * CIN * 9) + widx], v);
  wmix[gid] = f2bf(v);
}

// ============================================================================
// Kernel 3: x (fp32, [b][cin][h][w]) -> xT (bf16, [b][h][w][cin])
// cin-contiguous so the conv kernel's LDS fill is straight 128B copies.
// ============================================================================
__global__ void xcast_kernel(const float* __restrict__ x,
                             unsigned short* __restrict__ xT) {
  const int br  = blockIdx.x;            // b*64 + row
  const int b   = br >> 6, row = br & 63;
  const int tid = threadIdx.x;           // 256
  __shared__ unsigned short tile[CIN][W + 2];
  #pragma unroll
  for (int i = 0; i < 16; ++i) {
    int idx = i * 256 + tid;
    int cin = idx >> 6, col = idx & 63;
    tile[cin][col] = f2bf(x[(((size_t)b * CIN + cin) * H + row) * W + col]);
  }
  __syncthreads();
  #pragma unroll
  for (int i = 0; i < 16; ++i) {
    int idx = i * 256 + tid;
    int col = idx >> 6, cin = idx & 63;
    xT[((((size_t)b * H + row) * W) + col) * CIN + cin] = tile[cin][col];
  }
}

// ============================================================================
// Kernel 4: per-sample conv via bf16 WMMA (D = A x B + C, f32 accumulate).
// 3x3 conv = 9 shifted 1x1 GEMMs; K = Cin = 64 -> 2 WMMA k-steps of 32.
// Workgroup = 16 waves (512 thr), tile = 64 Cout x (4 rows x 64 cols), one batch.
// Wave (mg, rr): 16 Cout x 1 row x 4 col-tiles -> only 4 v8f accumulators,
// 9*2*4 = 72 v_wmma_f32_16x16x32_bf16 per wave. Observed ~80 VGPRs live ->
// __launch_bounds__(512, 8) caps at 128 VGPRs: two 16-wave workgroups per WGP
// (8 waves/SIMD, 114KB/320KB LDS) to hide the per-WMMA ds_load latency.
// Input tile staged via CDNA5 async global->LDS DMA (ASYNCcnt path).
// ============================================================================
__global__ __launch_bounds__(512, 8) void conv_kernel(
    const unsigned short* __restrict__ xT,
    const unsigned short* __restrict__ wmix,
    const float* __restrict__ bias_eff,
    float* __restrict__ out) {
  extern __shared__ unsigned short smem[];   // [6 rows][66 cols][72 cin-pitch] bf16
  const int yt  = blockIdx.x;                // 16 row-groups of 4 output rows
  const int b   = blockIdx.y;
  const int y0  = yt * 4;
  const int tid = threadIdx.x;

  const unsigned short* wb = wmix + (size_t)b * FRAGS_PER_B;
  __builtin_prefetch(wb + (size_t)(tid & 255) * 64, 0, 1); // warm L2 for A-fragments

  // ---- stage 6 input rows (y0..y0+5, clamped) into LDS, cin-contiguous ----
  // CDNA5 async DMA: global -> LDS directly, no VGPR round-trip (ASYNCcnt).
  {
    const unsigned short* xTb = xT + (size_t)b * H * W * CIN;   // SGPR base (GVS mode)
    const unsigned lds_base = __builtin_amdgcn_groupstaticsize(); // start of dynamic LDS
    for (int i = tid; i < 6 * LCOLS * 8; i += 512) {
      int pair = i >> 3, sub = i & 7;          // 8 x 16B per (row,col)
      int irow = pair / LCOLS, col = pair % LCOLS;
      int grow = min(y0 + irow, H - 1);
      int gcol = min(col, W - 1);
      unsigned voff    = (unsigned)(((grow * W + gcol) * CIN + sub * 8) * 2);
      unsigned ldsaddr = lds_base + (unsigned)((pair * LPITCH + sub * 8) * 2);
      asm volatile("global_load_async_to_lds_b128 %0, %1, %2"
                   :: "v"(ldsaddr), "v"(voff), "s"(xTb) : "memory");
    }
    asm volatile("s_wait_asynccnt 0" ::: "memory");
  }
  __syncthreads();

  const int wid   = tid >> 5;    // 0..15
  const int lane  = tid & 31;
  const int mg    = wid & 3;     // Cout group of 16
  const int rr    = wid >> 2;    // output row within the 4-row group (0..3)
  const int khalf = lane >> 4;
  const int nn    = lane & 15;

  v8f acc[4] = {};

  #pragma unroll
  for (int kh = 0; kh < 3; ++kh) {
    const int irow = rr + kh;
    #pragma unroll
    for (int kw = 0; kw < 3; ++kw) {
      #pragma unroll
      for (int ks = 0; ks < 2; ++ks) {
        const int pos = kh * 3 + kw;
        // A fragment (pre-laid-out in WMMA order): 32B per lane
        const v16bf a = *(const v16bf*)(wb + ((((pos * 2 + ks) * 4 + mg) * 32 + lane) << 4));
        #pragma unroll
        for (int ct = 0; ct < 4; ++ct) {
          // B fragment: 16 contiguous bf16 along cin at (irow, col=ct*16+nn+kw)
          const v16bf bm = *(const v16bf*)((const char*)smem +
              (((size_t)(irow * LCOLS + ct * 16 + nn + kw)) * LPITCH + ks * 32 + khalf * 16) * 2);
          acc[ct] = __builtin_amdgcn_wmma_f32_16x16x32_bf16(
              false, a, false, bm, (short)0, acc[ct], false, false);
        }
      }
    }
  }

  // ---- bias + masked store (valid region 62x62) ----
  const float4* bp = (const float4*)(bias_eff + b * COUT + mg * 16 + khalf * 8);
  const float4 b0 = bp[0], b1 = bp[1];
  const float bias8[8] = {b0.x, b0.y, b0.z, b0.w, b1.x, b1.y, b1.z, b1.w};

  const int y = y0 + rr;
  if (y < HO) {
    #pragma unroll
    for (int ct = 0; ct < 4; ++ct) {
      const int xcol = ct * 16 + nn;
      if (xcol >= WO) continue;
      #pragma unroll
      for (int j = 0; j < 8; ++j) {
        const int cout = mg * 16 + khalf * 8 + j;   // D layout: lanes 16-31 -> M = j+8
        out[(((size_t)b * COUT + cout) * HO + y) * WO + xcol] = acc[ct][j] + bias8[j];
      }
    }
  }
}

// ============================================================================
extern "C" void kernel_launch(void* const* d_in, const int* in_sizes, int n_in,
                              void* d_out, int out_size, void* d_ws, size_t ws_size,
                              hipStream_t stream) {
  const float* x   = (const float*)d_in[0];   // [32,64,64,64]
  const float* rv  = (const float*)d_in[1];   // [32,512]
  const float* cw  = (const float*)d_in[2];   // [10,64,64,3,3]
  const float* cb  = (const float*)d_in[3];   // [10,64]
  const float* emb = (const float*)d_in[4];   // [10,128]
  const float* rpw = (const float*)d_in[5];   // [128,512]
  const float* rpb = (const float*)d_in[6];   // [128]
  (void)d_in[7];                              // task: unused by reference output
  float* out = (float*)d_out;                 // [32,64,62,62] fp32

  char* ws = (char*)d_ws;
  float*          wroute  = (float*)(ws + 0);                       // 1280 B
  float*          biaseff = (float*)(ws + 2048);                    // 8192 B
  unsigned short* wmix    = (unsigned short*)(ws + 16384);          // 32*36864*2 B
  unsigned short* xT      = (unsigned short*)(ws + 16384 + (size_t)BATCH * FRAGS_PER_B * 2);

  routing_kernel<<<BATCH, 128, 0, stream>>>(rv, rpw, rpb, emb, cb, wroute, biaseff);
  mixw_kernel<<<(BATCH * FRAGS_PER_B) / 256, 256, 0, stream>>>(cw, wroute, wmix);
  xcast_kernel<<<BATCH * H, 256, 0, stream>>>(x, xT);
  conv_kernel<<<dim3(16, BATCH), 512, LDSX_BYTES, stream>>>(xT, wmix, biaseff, out);
}
